// GraghHopTransformer_84542136254920
// MI455X (gfx1250) — matrix operand order repaired and checked by reading, not compiled, not run on hardware
//
#include <hip/hip_runtime.h>
#include <hip/hip_bf16.h>

// ---------------------------------------------------------------------------
// CDNA5 (gfx1250) fused TGAT hop aggregation.
// Heavy K/V projection GEMMs run on v_wmma_f32_16x16x32_bf16.
// ---------------------------------------------------------------------------

typedef __attribute__((ext_vector_type(16))) __bf16 v16bf;
typedef __attribute__((ext_vector_type(8)))  float  v8f;

#define SKV 392   // kv_in LDS row stride (384 + 8 pad) in bf16 elems
#define SKO 264   // k/v output LDS row stride (256 + 8 pad) in bf16 elems

union BFragU {
    v16bf v;
    unsigned int u[8];
};

__device__ inline v8f zero8() {
    v8f z = {0.f, 0.f, 0.f, 0.f, 0.f, 0.f, 0.f, 0.f};
    return z;
}

// A fragment: 16x32 bf16 tile at (m0, k0) of the LDS kv_in buffer.
// ISA layout: lanes 0-15 hold rows M=0..15 with K base 0, lanes 16-31 same
// rows with K base 8; VGPR j (j<4): K = base+2j,+1; VGPR 4..7: K = base+16+...
__device__ inline v16bf load_a_frag(const __bf16* s_kv, int m0, int k0, int lane) {
    const int m     = m0 + (lane & 15);
    const int kbase = k0 + ((lane >> 4) << 3);
    const __bf16* rowp = s_kv + m * SKV + kbase;
    BFragU f;
#pragma unroll
    for (int j = 0; j < 4; ++j)
        f.u[j] = *reinterpret_cast<const unsigned int*>(rowp + 2 * j);
#pragma unroll
    for (int j = 0; j < 4; ++j)
        f.u[4 + j] = *reinterpret_cast<const unsigned int*>(rowp + 16 + 2 * j);
    return f.v;
}

// B fragment: 32x16 bf16 tile of a row-major [384 x 128] weight at (k0, n0).
// ISA layout: lane l holds column N=n0+(l&15); K base 8*(l>=16); elements
// 0..7: K = base+s, elements 8..15: K = base+16+(s-8) (same as A).
__device__ inline v16bf load_b_frag(const __bf16* W, int k0, int n0, int lane) {
    const int kbase = k0 + ((lane >> 4) << 3);
    const __bf16* p = W + (size_t)kbase * 128 + n0 + (lane & 15);
    v16bf f;
#pragma unroll
    for (int j = 0; j < 8; ++j) {
        f[j]     = p[(size_t)j * 128];
        f[8 + j] = p[(size_t)(16 + j) * 128];
    }
    return f;
}

__device__ inline v8f wmma_bf16(v16bf a, v16bf b, v8f c) {
    return __builtin_amdgcn_wmma_f32_16x16x32_bf16(
        /*neg_a=*/false, a, /*neg_b=*/false, b,
        /*c_mod=*/(short)0, c, /*reuse_a=*/false, /*reuse_b=*/false);
}

// ---------------------------------------------------------------------------
// prep: fp32 -> bf16 weight conversion + constant time encoding of t=0.
// ---------------------------------------------------------------------------
__global__ void prep_kernel(const float* __restrict__ Wk,
                            const float* __restrict__ Wv,
                            const float* __restrict__ time_b,
                            __bf16* __restrict__ wkb,
                            __bf16* __restrict__ wvb,
                            float* __restrict__ src_t) {
    int i = blockIdx.x * blockDim.x + threadIdx.x;
    const int n = 2 * 384 * 128;
    if (i < n) {
        wkb[i] = (__bf16)Wk[i];
        wvb[i] = (__bf16)Wv[i];
    }
    if (i < 128) src_t[i] = cosf(time_b[i]);
}

// ---------------------------------------------------------------------------
// Fused hop aggregation: one block per output row.
//   kv_in = [neigh_emb | time_enc(ts - etime) | edge_feat]   (40 x 384)
//   K = kv_in @ Wk ; V = kv_in @ Wv          (WMMA bf16)
//   2-head attention (dh=64) + Wo + ReLU FFN (fp32 VALU epilogue)
// ---------------------------------------------------------------------------
template <bool GATHER>
__global__ __launch_bounds__(256)
void hop_agg_kernel(const float* __restrict__ node_features,
                    const float* __restrict__ edge_features,
                    const float* __restrict__ time_w,
                    const float* __restrict__ time_b,
                    const float* __restrict__ src_t_vec,
                    const __bf16* __restrict__ Wk_l,
                    const __bf16* __restrict__ Wv_l,
                    const float* __restrict__ Wq_l,
                    const float* __restrict__ Wo_l,
                    const float* __restrict__ fc1_l,
                    const float* __restrict__ fc2_l,
                    const int* __restrict__ src_nodes,
                    const float* __restrict__ timestamps,
                    int ts_div,
                    const int* __restrict__ neigh,
                    const int* __restrict__ eidx,
                    const float* __restrict__ etimes,
                    const float* __restrict__ emb_in,
                    float* __restrict__ out) {
    __shared__ __bf16 s_kv[48 * SKV];   // kv_in tile (rows 40..47 zero)
    __shared__ __bf16 s_ko[40 * SKO];   // [K | V] projection output
    __shared__ float  s_src[256];       // [src features | src time enc]
    __shared__ float  s_q[128];
    __shared__ float  s_sc[2][40];
    __shared__ float  s_vA[128];
    __shared__ float  s_vB[128];
    __shared__ int    s_nid[40];
    __shared__ int    s_eid[40];
    __shared__ float  s_dt[40];

    const int tid = threadIdx.x;
    const int r = blockIdx.x;
    const float ts = timestamps[r / ts_div];

    if (tid < 40) {
        s_nid[tid] = neigh[(size_t)r * 40 + tid];
        s_eid[tid] = eidx[(size_t)r * 40 + tid];
        s_dt[tid]  = ts - etimes[(size_t)r * 40 + tid];
    }
    if (tid < 128) {
        const int sn = src_nodes[r];
        s_src[tid]       = node_features[(size_t)sn * 128 + tid];
        s_src[128 + tid] = src_t_vec[tid];
    }
    __syncthreads();

    // ---- build kv_in tile in bf16 --------------------------------------
    for (int idx = tid; idx < 48 * 384; idx += 256) {
        const int row = idx / 384;
        const int c = idx - row * 384;
        float val = 0.f;
        if (row < 40) {
            if (c < 128) {
                val = GATHER ? node_features[(size_t)s_nid[row] * 128 + c]
                             : emb_in[((size_t)r * 40 + row) * 128 + c];
            } else if (c < 256) {
                const int j = c - 128;
                val = cosf(s_dt[row] * time_w[j] + time_b[j]);
            } else {
                val = edge_features[(size_t)s_eid[row] * 128 + (c - 256)];
            }
        }
        s_kv[row * SKV + c] = (__bf16)val;
    }
    __syncthreads();

    // ---- WMMA GEMM: [48 x 384] @ [384 x (128K | 128V)] -----------------
    const int wave = tid >> 5;
    const int lane = tid & 31;

    v8f acc[2][3];
#pragma unroll
    for (int t = 0; t < 2; ++t)
#pragma unroll
        for (int m = 0; m < 3; ++m) acc[t][m] = zero8();

#pragma unroll
    for (int kt = 0; kt < 12; ++kt) {
        const int k0 = kt * 32;
        v16bf a0 = load_a_frag(s_kv, 0, k0, lane);
        v16bf a1 = load_a_frag(s_kv, 16, k0, lane);
        v16bf a2 = load_a_frag(s_kv, 32, k0, lane);
#pragma unroll
        for (int t = 0; t < 2; ++t) {
            const int ncol = (wave * 2 + t) * 16;
            const __bf16* W = (ncol < 128) ? Wk_l : Wv_l;
            v16bf b = load_b_frag(W, k0, ncol & 127, lane);
            acc[t][0] = wmma_bf16(a0, b, acc[t][0]);
            acc[t][1] = wmma_bf16(a1, b, acc[t][1]);
            acc[t][2] = wmma_bf16(a2, b, acc[t][2]);
        }
    }

    // D layout: VGPR v -> M = m0 + v + 8*(lane>=16), N = n0 + (lane&15)
    {
        const int half = lane >> 4;
        const int colL = lane & 15;
#pragma unroll
        for (int t = 0; t < 2; ++t) {
            const int n0 = (wave * 2 + t) * 16;
#pragma unroll
            for (int mt = 0; mt < 3; ++mt) {
#pragma unroll
                for (int v = 0; v < 8; ++v) {
                    const int row = mt * 16 + v + 8 * half;
                    if (row < 40)
                        s_ko[row * SKO + n0 + colL] = (__bf16)acc[t][mt][v];
                }
            }
        }
    }
    __syncthreads();

    // ---- q = [src | src_t] @ Wq ----------------------------------------
    if (tid < 128) {
        float s = 0.f;
        for (int i = 0; i < 256; ++i) s += s_src[i] * Wq_l[i * 128 + tid];
        s_q[tid] = s;
    }
    __syncthreads();

    // ---- scores = q . k / sqrt(64), masked -----------------------------
    if (tid < 80) {
        const int h = tid / 40;
        const int n = tid % 40;
        const __bf16* kp = s_ko + n * SKO + h * 64;
        float s = 0.f;
        for (int j = 0; j < 64; ++j) s += s_q[h * 64 + j] * (float)kp[j];
        s *= 0.125f;
        if (s_nid[n] == 0) s = -1e9f;
        s_sc[h][n] = s;
    }
    __syncthreads();

    // ---- softmax over 40 neighbors (per head) --------------------------
    if (tid < 2) {
        float m = -3.4e38f;
        for (int n = 0; n < 40; ++n) m = fmaxf(m, s_sc[tid][n]);
        float sum = 0.f;
        for (int n = 0; n < 40; ++n) {
            const float e = expf(s_sc[tid][n] - m);
            s_sc[tid][n] = e;
            sum += e;
        }
        const float inv = 1.f / sum;
        for (int n = 0; n < 40; ++n) s_sc[tid][n] *= inv;
    }
    __syncthreads();

    // ---- out = attn @ V -------------------------------------------------
    if (tid < 128) {
        const int h = tid >> 6;
        float s = 0.f;
        for (int n = 0; n < 40; ++n)
            s += s_sc[h][n] * (float)s_ko[n * SKO + 128 + tid];
        s_vA[tid] = s;
    }
    __syncthreads();

    // ---- o2 = out @ Wo --------------------------------------------------
    if (tid < 128) {
        float s = 0.f;
        for (int i = 0; i < 128; ++i) s += s_vA[i] * Wo_l[i * 128 + tid];
        s_vB[tid] = s;
    }
    __syncthreads();

    // ---- h1 = relu([o2 | src] @ fc1) ------------------------------------
    if (tid < 128) {
        float s = 0.f;
        for (int i = 0; i < 128; ++i) s += s_vB[i] * fc1_l[i * 128 + tid];
        for (int i = 0; i < 128; ++i) s += s_src[i] * fc1_l[(128 + i) * 128 + tid];
        s_vA[tid] = fmaxf(s, 0.f);
    }
    __syncthreads();

    // ---- h2 = h1 @ fc2 -> output ---------------------------------------
    if (tid < 128) {
        float s = 0.f;
        for (int i = 0; i < 128; ++i) s += s_vA[i] * fc2_l[i * 128 + tid];
        out[(size_t)r * 128 + tid] = s;
    }
}

// ---------------------------------------------------------------------------
extern "C" void kernel_launch(void* const* d_in, const int* in_sizes, int n_in,
                              void* d_out, int out_size, void* d_ws, size_t ws_size,
                              hipStream_t stream) {
    const float* node_features = (const float*)d_in[0];
    const float* edge_features = (const float*)d_in[1];
    const float* time_w        = (const float*)d_in[2];
    const float* time_b        = (const float*)d_in[3];
    const float* Wq            = (const float*)d_in[4];
    const float* Wk            = (const float*)d_in[5];
    const float* Wv            = (const float*)d_in[6];
    const float* Wo            = (const float*)d_in[7];
    const float* fc1           = (const float*)d_in[8];
    const float* fc2           = (const float*)d_in[9];
    const int*   source_nodes  = (const int*)d_in[10];
    const float* timestamps    = (const float*)d_in[11];
    const int*   neigh2        = (const int*)d_in[12];
    const int*   eidx2         = (const int*)d_in[13];
    const float* etimes2       = (const float*)d_in[14];
    const int*   neigh1        = (const int*)d_in[15];
    const int*   eidx1         = (const int*)d_in[16];
    const float* etimes1       = (const float*)d_in[17];

    // Workspace layout (bytes):
    //   [0)        bf16 Wk   2*384*128 -> 196608
    //   [196608)   bf16 Wv   2*384*128 -> 196608
    //   [393216)   f32  src_t (cos(time_b)) -> 512
    //   [393728)   f32  emb1  12000*128 -> 6144000
    char* ws = (char*)d_ws;
    __bf16* wkb   = (__bf16*)(ws);
    __bf16* wvb   = (__bf16*)(ws + 196608);
    float*  src_t = (float*)(ws + 393216);
    float*  emb1  = (float*)(ws + 393728);

    prep_kernel<<<384, 256, 0, stream>>>(Wk, Wv, time_b, wkb, wvb, src_t);

    // Level 1 (layer 0): 12000 aggregation rows (2-hop neighbors of sources).
    hop_agg_kernel<true><<<12000, 256, 0, stream>>>(
        node_features, edge_features, time_w, time_b, src_t,
        wkb, wvb, Wq, Wo, fc1, fc2,
        /*src_nodes=*/neigh2, timestamps, /*ts_div=*/40,
        neigh1, eidx1, etimes1, /*emb_in=*/nullptr, emb1);

    // Level 2 (layer 1): 300 source rows, neighbor embeddings from emb1.
    hop_agg_kernel<false><<<300, 256, 0, stream>>>(
        node_features, edge_features, time_w, time_b, src_t,
        wkb + 384 * 128, wvb + 384 * 128,
        Wq + 256 * 128, Wo + 128 * 128,
        fc1 + 256 * 128, fc2 + 128 * 128,
        source_nodes, timestamps, /*ts_div=*/1,
        neigh2, eidx2, etimes2, emb1, (float*)d_out);
}
